// CompanyOperationEvaluation_249108103519
// MI455X (gfx1250) — matrix operand order repaired and checked by reading, not compiled, not run on hardware
//
#include <hip/hip_runtime.h>

// ---------------------------------------------------------------------------
// Types for CDNA5 WMMA (wave32)
// ---------------------------------------------------------------------------
typedef _Float16 half_t;
typedef _Float16 v4h  __attribute__((ext_vector_type(4)));
typedef _Float16 v8h  __attribute__((ext_vector_type(8)));
typedef _Float16 v16h __attribute__((ext_vector_type(16)));
typedef float    v8f  __attribute__((ext_vector_type(8)));
typedef int      i32x4 __attribute__((__vector_size__(16)));

static constexpr int Bn  = 16384;
static constexpr int Fd  = 256;
static constexpr int Dd  = 128;
static constexpr int H0  = 512;
static constexpr int H1  = 256;
static constexpr int OUTC = 8;

#define AS1 __attribute__((address_space(1)))
#define AS3 __attribute__((address_space(3)))

#if __has_builtin(__builtin_amdgcn_global_load_async_to_lds_b128) && \
    __has_builtin(__builtin_amdgcn_s_wait_asynccnt)
#define ASYNC_LDS 1
#else
#define ASYNC_LDS 0
#endif

__device__ __forceinline__ v16h pack16(v8h lo, v8h hi) {
    v16h r;
#pragma unroll
    for (int i = 0; i < 8; ++i) { r[i] = lo[i]; r[i + 8] = hi[i]; }
    return r;
}

#if ASYNC_LDS
__device__ __forceinline__ void async_copy16(const void* g, void* l) {
    __builtin_amdgcn_global_load_async_to_lds_b128(
        (AS1 i32x4*)g, (AS3 i32x4*)l, 0, 0);
}
#endif

// ---------------------------------------------------------------------------
// Pre-swizzle W (K x N, row-major f32) into the contiguous-k B-fragment order used here
// (f16), blocked [N/16][K/32][512 halves]. Within a block:
//   plane 0 (halves 0..255): lane*8+h8 -> b[h8]     (h = h8)
//   plane 1 (halves 256..511): lane*8+h8 -> b[8+h8] (h = 8+h8)
// B layout used (16x16x32 f16): lane<16 : col=lane,    K = k32*32 + h
//                              lane>=16: col=lane-16, K = k32*32 + 16 + h
// Plane split => one ds_load_b128 across the wave touches 128 consecutive
// dwords (minimal bank traffic).
// ---------------------------------------------------------------------------
template<int K, int N>
__global__ __launch_bounds__(256) void swizzle_w_kernel(
        const float* __restrict__ W, half_t* __restrict__ Wsw) {
    const int idx = blockIdx.x * 256 + threadIdx.x;
    constexpr int K32 = K / 32;
    constexpr int total = (N / 16) * K32 * 512;
    if (idx >= total) return;
    const int pos  = idx & 511;
    const int k32  = (idx >> 9) % K32;
    const int n16  = idx / (512 * K32);
    const int p    = pos >> 8;          // plane
    const int lane = (pos >> 3) & 31;
    const int h    = p * 8 + (pos & 7); // half index 0..15 within B operand
    const int col  = n16 * 16 + (lane & 15);
    const int kk   = k32 * 32 + ((lane < 16) ? h : 16 + h);
    Wsw[idx] = (half_t)W[(size_t)kk * N + col];
}

// ---------------------------------------------------------------------------
// Fused GEMM + bias + ReLU on WMMA f16->f32.
//   A: (B x K) row-major, f32 or f16 (AF32 flag); Bsw: pre-swizzled weights.
//   Out: f16 row-major, leading dim ldo (stage-3 writes into the concat).
// WG: 256 threads = 8 waves, 128 rows. Dynamic LDS:
//   ldsA: 128 x (K+8) halves  (pad kills the %64-dword row-stride conflict)
//   ldsB: 128*K halves        (current 128-column chunk, swizzled blocks)
// ---------------------------------------------------------------------------
template<int K, int N, bool AF32>
__global__ __launch_bounds__(256) void gemm_relu_wmma_kernel(
        const void* __restrict__ Aptr, const half_t* __restrict__ Bsw,
        const float* __restrict__ bias, half_t* __restrict__ Out, int ldo) {
    extern __shared__ char smem[];
    constexpr int LDA = K + 8;
    half_t* ldsA = (half_t*)smem;
    half_t* ldsB = (half_t*)smem + 128 * LDA;

    const int tid     = threadIdx.x;
    const int wave    = tid >> 5;
    const int lane    = tid & 31;
    const int rowBase = blockIdx.x * 128;
    constexpr int K32  = K / 32;
    constexpr int LOGK = (K == 128) ? 7 : ((K == 256) ? 8 : 9);

    // ---- stage A tile (128 x K) into LDS as f16 (padded rows) ----
    if (AF32) {
        const float* A = (const float*)Aptr + (size_t)rowBase * K;
        for (int i = tid * 4; i < 128 * K; i += 256 * 4) {
            const int row = i >> LOGK, k = i & (K - 1);
            float4 v = *(const float4*)(A + i);
            v4h h; h[0] = (half_t)v.x; h[1] = (half_t)v.y;
                   h[2] = (half_t)v.z; h[3] = (half_t)v.w;
            *(v4h*)(ldsA + row * LDA + k) = h;
        }
    } else {
        const half_t* A = (const half_t*)Aptr + (size_t)rowBase * K;
        for (int i = tid * 8; i < 128 * K; i += 256 * 8) {
            const int row = i >> LOGK, k = i & (K - 1);
#if ASYNC_LDS
            async_copy16(A + i, ldsA + row * LDA + k);
#else
            *(v8h*)(ldsA + row * LDA + k) = *(const v8h*)(A + i);
#endif
        }
    }

    const int arow = wave * 16 + (lane & 15);
    const int koff = (lane < 16) ? 0 : 16;

    for (int nc = 0; nc < N; nc += 128) {
        __syncthreads();   // A ready (1st iter) / previous chunk done
        // ---- stage B panel (8 n16-blocks, contiguous in swizzled layout) ----
        const half_t* Bp = Bsw + (size_t)(nc / 16) * K32 * 512;
        constexpr int panelHalves = 8 * K32 * 512;
        for (int i = tid * 8; i < panelHalves; i += 256 * 8) {
#if ASYNC_LDS
            async_copy16(Bp + i, ldsB + i);
#else
            *(v8h*)(ldsB + i) = *(const v8h*)(Bp + i);
#endif
        }
        if (nc + 128 < N)
            __builtin_prefetch(Bsw + (size_t)((nc + 128) / 16) * K32 * 512, 0, 0);
#if ASYNC_LDS
        __builtin_amdgcn_s_wait_asynccnt(0);
#endif
        __syncthreads();

        v8f acc[8];
#pragma unroll
        for (int nb = 0; nb < 8; ++nb)
#pragma unroll
            for (int r = 0; r < 8; ++r) acc[nb][r] = 0.0f;

        for (int k32 = 0; k32 < K32; ++k32) {
            const half_t* ap = ldsA + (size_t)arow * LDA + k32 * 32 + koff;
            v16h a = pack16(*(const v8h*)ap, *(const v8h*)(ap + 8));
#pragma unroll
            for (int nb = 0; nb < 8; ++nb) {
                const half_t* bb = ldsB + ((nb * K32 + k32) << 9) + lane * 8;
                v16h b = pack16(*(const v8h*)bb, *(const v8h*)(bb + 256));
                acc[nb] = __builtin_amdgcn_wmma_f32_16x16x32_f16(
                    false, a, false, b, (short)0, acc[nb], false, false);
            }
        }

        // ---- bias + ReLU + f16 store (C/D: VGPR r, lane l ->
        //      M = r + (l<16?0:8), N = l&15) ----
        const int n0   = nc + (lane & 15);
        const int mrow = rowBase + wave * 16 + ((lane < 16) ? 0 : 8);
#pragma unroll
        for (int nb = 0; nb < 8; ++nb) {
            const int n  = n0 + nb * 16;
            const float bs = bias[n];
#pragma unroll
            for (int r = 0; r < 8; ++r) {
                float v = acc[nb][r] + bs;
                v = (v > 0.0f) ? v : 0.0f;
                Out[(size_t)(mrow + r) * ldo + n] = (half_t)v;
            }
        }
    }
}

constexpr size_t gemm_smem(int K) { return (size_t)(128 * (K + 8) + 128 * K) * 2; }

// ---------------------------------------------------------------------------
// Fused embedding gather + 2x cross_compress; writes e2 (f16) into
// high[:, 128:256]. One wave per row, 4 f32 elements per lane (D=128).
// ---------------------------------------------------------------------------
__device__ __forceinline__ float wave_sum(float v) {
#pragma unroll
    for (int m = 16; m; m >>= 1) v += __shfl_xor(v, m, 32);
    return v;
}

__global__ __launch_bounds__(256) void cross_compress_kernel(
        const float* __restrict__ head_tab, const float* __restrict__ ent_tab,
        const int* __restrict__ ent_idx,
        const float* __restrict__ w_cf, const float* __restrict__ w_fc,
        const float* __restrict__ w_ef, const float* __restrict__ w_fe,
        const float* __restrict__ b_c,  const float* __restrict__ b_e,
        half_t* __restrict__ high) {
    const int wave = threadIdx.x >> 5;
    const int lane = threadIdx.x & 31;
    const int row  = blockIdx.x * 8 + wave;
    const int id   = ent_idx[row];

    float4 f   = ((const float4*)(head_tab + (size_t)id * Dd))[lane];
    float4 e   = ((const float4*)(ent_tab  + (size_t)id * Dd))[lane];
    float4 wcf = ((const float4*)w_cf)[lane];
    float4 wfc = ((const float4*)w_fc)[lane];
    float4 wef = ((const float4*)w_ef)[lane];
    float4 wfe = ((const float4*)w_fe)[lane];
    const float bc = b_c[0], be = b_e[0];

#pragma unroll
    for (int it = 0; it < 2; ++it) {
        float s_ecf = wave_sum(e.x * wcf.x + e.y * wcf.y + e.z * wcf.z + e.w * wcf.w);
        float s_ffc = wave_sum(f.x * wfc.x + f.y * wfc.y + f.z * wfc.z + f.w * wfc.w);
        float s_eef = wave_sum(e.x * wef.x + e.y * wef.y + e.z * wef.z + e.w * wef.w);
        float s_ffe = wave_sum(f.x * wfe.x + f.y * wfe.y + f.z * wfe.z + f.w * wfe.w);
        float4 nf, ne;
        nf.x = f.x * s_ecf + e.x * s_ffc + bc;  ne.x = f.x * s_eef + e.x * s_ffe + be;
        nf.y = f.y * s_ecf + e.y * s_ffc + bc;  ne.y = f.y * s_eef + e.y * s_ffe + be;
        nf.z = f.z * s_ecf + e.z * s_ffc + bc;  ne.z = f.z * s_eef + e.z * s_ffe + be;
        nf.w = f.w * s_ecf + e.w * s_ffc + bc;  ne.w = f.w * s_eef + e.w * s_ffe + be;
        f = nf; e = ne;
    }
    v4h o; o[0] = (half_t)e.x; o[1] = (half_t)e.y; o[2] = (half_t)e.z; o[3] = (half_t)e.w;
    *(v4h*)(high + (size_t)row * 256 + 128 + lane * 4) = o;
}

// ---------------------------------------------------------------------------
// Final head: relu(x1 @ W2 + b2) -> softmax -> prob (f32), plus target copy.
// ---------------------------------------------------------------------------
__global__ __launch_bounds__(256) void head_softmax_kernel(
        const half_t* __restrict__ x1, const float* __restrict__ W2,
        const float* __restrict__ b2, const int* __restrict__ target,
        float* __restrict__ out) {
    __shared__ float w2s[H1 * OUTC];
    for (int i = threadIdx.x; i < H1 * OUTC; i += 256) w2s[i] = W2[i];
    __syncthreads();

    const int wave = threadIdx.x >> 5;
    const int lane = threadIdx.x & 31;
    const int row  = blockIdx.x * 8 + wave;

    v8h xv = *(const v8h*)(x1 + (size_t)row * H1 + lane * 8);
    float acc[OUTC];
#pragma unroll
    for (int o = 0; o < OUTC; ++o) acc[o] = 0.0f;
#pragma unroll
    for (int j = 0; j < 8; ++j) {
        const float xf = (float)xv[j];
        const int k = lane * 8 + j;
#pragma unroll
        for (int o = 0; o < OUTC; ++o) acc[o] += xf * w2s[k * OUTC + o];
    }
#pragma unroll
    for (int o = 0; o < OUTC; ++o) acc[o] = wave_sum(acc[o]);

    if (lane == 0) {
        float v[OUTC], mx = -1e30f;
#pragma unroll
        for (int o = 0; o < OUTC; ++o) {
            float t = acc[o] + b2[o];
            v[o] = (t > 0.0f) ? t : 0.0f;
            mx = (v[o] > mx) ? v[o] : mx;
        }
        float s = 0.0f;
#pragma unroll
        for (int o = 0; o < OUTC; ++o) { v[o] = __expf(v[o] - mx); s += v[o]; }
        const float inv = 1.0f / s;
#pragma unroll
        for (int o = 0; o < OUTC; ++o) out[(size_t)row * OUTC + o] = v[o] * inv;
        out[(size_t)Bn * OUTC + row] = (float)target[row];
    }
}

// ---------------------------------------------------------------------------
// Launcher
// ---------------------------------------------------------------------------
extern "C" void kernel_launch(void* const* d_in, const int* in_sizes, int n_in,
                              void* d_out, int out_size, void* d_ws, size_t ws_size,
                              hipStream_t stream) {
    (void)in_sizes; (void)n_in; (void)out_size; (void)ws_size;

    const float* features = (const float*)d_in[0];
    const int*   ent_idx  = (const int*)d_in[1];
    const int*   target   = (const int*)d_in[2];
    const float* Wf = (const float*)d_in[3];  const float* bf = (const float*)d_in[4];
    const float* Wu = (const float*)d_in[5];  const float* bu = (const float*)d_in[6];
    const float* w_cf = (const float*)d_in[7];  const float* w_fc = (const float*)d_in[8];
    const float* w_ef = (const float*)d_in[9];  const float* w_fe = (const float*)d_in[10];
    const float* b_c = (const float*)d_in[11];  const float* b_e = (const float*)d_in[12];
    const float* head_tab = (const float*)d_in[13];
    const float* ent_tab  = (const float*)d_in[14];
    const float* W0 = (const float*)d_in[15]; const float* b0 = (const float*)d_in[16];
    const float* W1 = (const float*)d_in[17]; const float* b1 = (const float*)d_in[18];
    const float* W2 = (const float*)d_in[19]; const float* b2 = (const float*)d_in[20];

    // ---- workspace carve-up (256B aligned) ----
    char* ws = (char*)d_ws;
    size_t off = 0;
    auto alloc = [&](size_t bytes) -> void* {
        void* p = ws + off;
        off = (off + bytes + 255) & ~(size_t)255;
        return p;
    };
    half_t* Wf_sw = (half_t*)alloc((size_t)Fd * Dd * 2);
    half_t* Wu_sw = (half_t*)alloc((size_t)Dd * Dd * 2);
    half_t* W0_sw = (half_t*)alloc((size_t)(2 * Dd) * H0 * 2);
    half_t* W1_sw = (half_t*)alloc((size_t)H0 * H1 * 2);
    half_t* cf0   = (half_t*)alloc((size_t)Bn * Dd * 2);
    half_t* cf1   = (half_t*)alloc((size_t)Bn * Dd * 2);
    half_t* high  = (half_t*)alloc((size_t)Bn * (2 * Dd) * 2);
    half_t* x0    = (half_t*)alloc((size_t)Bn * H0 * 2);
    half_t* x1    = (half_t*)alloc((size_t)Bn * H1 * 2);

    // ---- weight pre-swizzle to WMMA B-fragment order (f16) ----
    swizzle_w_kernel<Fd, Dd><<<(Dd/16)*(Fd/32)*2, 256, 0, stream>>>(Wf, Wf_sw);
    swizzle_w_kernel<Dd, Dd><<<(Dd/16)*(Dd/32)*2, 256, 0, stream>>>(Wu, Wu_sw);
    swizzle_w_kernel<2*Dd, H0><<<(H0/16)*((2*Dd)/32)*2, 256, 0, stream>>>(W0, W0_sw);
    swizzle_w_kernel<H0, H1><<<(H1/16)*(H0/32)*2, 256, 0, stream>>>(W1, W1_sw);

    const int gemmGrid = Bn / 128;

    // cf0 = relu(features @ Wf + bf)          K=256, N=128, A=f32
    gemm_relu_wmma_kernel<Fd, Dd, true><<<gemmGrid, 256, gemm_smem(Fd), stream>>>(
        features, Wf_sw, bf, cf0, Dd);
    // cf1 = relu(cf0 @ Wu + bu)               K=128, N=128
    gemm_relu_wmma_kernel<Dd, Dd, false><<<gemmGrid, 256, gemm_smem(Dd), stream>>>(
        cf0, Wu_sw, bu, cf1, Dd);
    // high[:, :128] = relu(cf1 @ Wu + bu)
    gemm_relu_wmma_kernel<Dd, Dd, false><<<gemmGrid, 256, gemm_smem(Dd), stream>>>(
        cf1, Wu_sw, bu, high, 2 * Dd);
    // high[:, 128:256] = cross_compress^2 entity branch
    cross_compress_kernel<<<Bn / 8, 256, 0, stream>>>(
        head_tab, ent_tab, ent_idx, w_cf, w_fc, w_ef, w_fe, b_c, b_e, high);
    // x0 = relu(high @ W0 + b0)               K=256, N=512
    gemm_relu_wmma_kernel<2 * Dd, H0, false><<<gemmGrid, 256, gemm_smem(2 * Dd), stream>>>(
        high, W0_sw, b0, x0, H0);
    // x1 = relu(x0 @ W1 + b1)                 K=512, N=256
    gemm_relu_wmma_kernel<H0, H1, false><<<gemmGrid, 256, gemm_smem(H0), stream>>>(
        x0, W1_sw, b1, x1, H1);
    // prob + target
    head_softmax_kernel<<<Bn / 8, 256, 0, stream>>>(x1, W2, b2, target, (float*)d_out);
}
